// InnerAttention_25572235280559
// MI455X (gfx1250) — matrix operand order repaired and checked by reading, not compiled, not run on hardware
//
#include <hip/hip_runtime.h>
#include <stdint.h>

#define B_ 8
#define M_ 2048
#define N_ 2048
#define D_ 1024

typedef __attribute__((ext_vector_type(8)))  float        v8f;
typedef __attribute__((ext_vector_type(16))) __bf16       v16bf;
typedef __attribute__((ext_vector_type(4)))  unsigned int v4u;

union Frag {
  v16bf bf;
  v4u   q[2];   // 32 bytes: two b128 halves
};

__device__ inline unsigned short f2bf(float f) {
  unsigned int u = __float_as_uint(f);
  unsigned int r = u + 0x7FFFu + ((u >> 16) & 1u);   // round-to-nearest-even
  return (unsigned short)(r >> 16);
}

__device__ inline unsigned int pack2(unsigned short lo, unsigned short hi) {
  return (unsigned int)lo | ((unsigned int)hi << 16);
}

__device__ inline v8f vzero8() {
  v8f r;
  #pragma unroll
  for (int i = 0; i < 8; ++i) r[i] = 0.f;
  return r;
}

__device__ inline v8f wmma_bf16(const Frag& a, const Frag& b, v8f c) {
  return __builtin_amdgcn_wmma_f32_16x16x32_bf16(
      /*neg_a=*/false, a.bf, /*neg_b=*/false, b.bf,
      /*c_mod=*/(short)0, c, /*reuse_a=*/false, /*reuse_b=*/false);
}

// ---------------------------------------------------------------------------
// Kernel 0: fp32 -> bf16 conversion (vectorized: float4 in, uint2 out)
// ---------------------------------------------------------------------------
__global__ void cvt_bf16_kernel(const float* __restrict__ src,
                                unsigned short* __restrict__ dst, int n4) {
  int i = blockIdx.x * blockDim.x + threadIdx.x;
  if (i >= n4) return;
  const float4 v = ((const float4*)src)[i];
  uint2 o;
  o.x = pack2(f2bf(v.x), f2bf(v.y));
  o.y = pack2(f2bf(v.z), f2bf(v.w));
  ((uint2*)dst)[i] = o;
}

// ---------------------------------------------------------------------------
// Kernel 1: x[b,m,d] f32 -> xT[b,d,m] bf16 (tiled LDS transpose, 32x32)
// ---------------------------------------------------------------------------
__global__ void __launch_bounds__(256)
transpose_bf16_kernel(const float* __restrict__ x, unsigned short* __restrict__ xT) {
  __shared__ float tile[32][33];
  const int b  = blockIdx.z;
  const int m0 = blockIdx.y * 32;
  const int d0 = blockIdx.x * 32;
  const int tx = threadIdx.x & 31;
  const int ty = threadIdx.x >> 5;   // 0..7
  const float* xr = x + ((size_t)b * M_ + m0) * D_ + d0;
  #pragma unroll
  for (int r = 0; r < 4; ++r)
    tile[ty + r * 8][tx] = xr[(size_t)(ty + r * 8) * D_ + tx];
  __syncthreads();
  unsigned short* o = xT + ((size_t)b * D_ + d0) * M_ + m0;
  #pragma unroll
  for (int r = 0; r < 4; ++r)
    o[(size_t)(ty + r * 8) * M_ + tx] = f2bf(tile[tx][ty + r * 8]);
}

// ---------------------------------------------------------------------------
// Kernel 2: row stats.  Block = (b, 64 m-rows); per n-iteration covers 128 n.
// 8 waves tiled 2(m-pair) x 4(n-pair); each wave register-blocks 2x2 tiles
// (32m x 32n) so A/B fragments are each reused twice: 2 b128 loads per WMMA.
// ---------------------------------------------------------------------------
__global__ void __launch_bounds__(256)
stats_kernel(const unsigned short* __restrict__ xb,
             const unsigned short* __restrict__ yb,
             float* __restrict__ rowmax, float* __restrict__ rowsum) {
  __shared__ float mxbuf[4][64];
  __shared__ float smbuf[4][64];
  const int b    = blockIdx.y;
  const int m0   = blockIdx.x * 64;
  const int tid  = threadIdx.x;
  const int wave = tid >> 5, lane = tid & 31;
  const int mi2  = wave >> 2;   // 0..1 : which 32-row m block
  const int nj2  = wave & 3;    // 0..3 : which 32-col n block within 128 strip
  const int h    = lane >> 4, ln = lane & 15;

  const unsigned short* a0row =
      xb + ((size_t)b * M_ + m0 + mi2 * 32 + ln) * D_;
  const unsigned short* a1row = a0row + (size_t)16 * D_;

  float mx[2][8], sm[2][8];
  #pragma unroll
  for (int s = 0; s < 2; ++s)
    #pragma unroll
    for (int v = 0; v < 8; ++v) { mx[s][v] = -1e30f; sm[s][v] = 0.f; }

  for (int n0 = 0; n0 < N_; n0 += 128) {
    const unsigned short* b0row =
        yb + ((size_t)b * N_ + n0 + nj2 * 32 + ln) * D_;
    const unsigned short* b1row = b0row + (size_t)16 * D_;
    if (n0 + 128 < N_) {           // warm L2/L0 for next strip's B rows
      __builtin_prefetch(b0row + (size_t)128 * D_, 0, 2);
      __builtin_prefetch(b1row + (size_t)128 * D_, 0, 2);
    }
    v8f acc[2][2];
    acc[0][0] = vzero8(); acc[0][1] = vzero8();
    acc[1][0] = vzero8(); acc[1][1] = vzero8();
    #pragma unroll 2
    for (int k0 = 0; k0 < D_; k0 += 32) {
      Frag a0, a1, bm0, bm1;
      a0.q[0]  = *(const v4u*)(a0row + k0 + h * 8);
      a0.q[1]  = *(const v4u*)(a0row + k0 + 16 + h * 8);
      a1.q[0]  = *(const v4u*)(a1row + k0 + h * 8);
      a1.q[1]  = *(const v4u*)(a1row + k0 + 16 + h * 8);
      bm0.q[0] = *(const v4u*)(b0row + k0 + h * 8);
      bm0.q[1] = *(const v4u*)(b0row + k0 + 16 + h * 8);
      bm1.q[0] = *(const v4u*)(b1row + k0 + h * 8);
      bm1.q[1] = *(const v4u*)(b1row + k0 + 16 + h * 8);
      acc[0][0] = wmma_bf16(a0, bm0, acc[0][0]);
      acc[0][1] = wmma_bf16(a0, bm1, acc[0][1]);
      acc[1][0] = wmma_bf16(a1, bm0, acc[1][0]);
      acc[1][1] = wmma_bf16(a1, bm1, acc[1][1]);
    }
    // online softmax update over the 32 n-columns this wave just produced
    #pragma unroll
    for (int s = 0; s < 2; ++s) {
      #pragma unroll
      for (int v = 0; v < 8; ++v) {
        float t0 = acc[s][0][v];
        float t1 = acc[s][1][v];
        float t  = fmaxf(t0, t1);
        for (int off = 8; off; off >>= 1) t = fmaxf(t, __shfl_xor(t, off, 32));
        float nm = fmaxf(mx[s][v], t);
        float p  = __expf(t0 - nm) + __expf(t1 - nm);
        for (int off = 8; off; off >>= 1) p += __shfl_xor(p, off, 32);
        sm[s][v] = sm[s][v] * __expf(mx[s][v] - nm) + p;
        mx[s][v] = nm;
      }
    }
  }
  if (ln == 0) {
    #pragma unroll
    for (int s = 0; s < 2; ++s)
      #pragma unroll
      for (int v = 0; v < 8; ++v) {
        int r = mi2 * 32 + s * 16 + v + 8 * h;
        mxbuf[nj2][r] = mx[s][v];
        smbuf[nj2][r] = sm[s][v];
      }
  }
  __syncthreads();
  if (tid < 64) {
    float gm = mxbuf[0][tid];
    for (int j = 1; j < 4; ++j) gm = fmaxf(gm, mxbuf[j][tid]);
    float gs = 0.f;
    for (int j = 0; j < 4; ++j) gs += smbuf[j][tid] * __expf(mxbuf[j][tid] - gm);
    rowmax[(size_t)b * M_ + m0 + tid] = gm;
    rowsum[(size_t)b * M_ + m0 + tid] = gs;
  }
}

// ---------------------------------------------------------------------------
// Kernel 3: out[b,n,d] = sum_m P[m,n] * x[m,d].
// Block = (b, 16 n-cols). Wave w owns d in [w*128, w*128+128) -> 8 accum tiles.
// Per 32-wide m chunk: waves split the D reduction of the e recompute,
// reduce partials in LDS, exponentiate with rowmax/rowsum, stage P^T in LDS
// in bf16 A-fragment layout, then WMMA against xT.
// ---------------------------------------------------------------------------
__global__ void __launch_bounds__(256)
out_kernel(const unsigned short* __restrict__ xb,
           const unsigned short* __restrict__ yb,
           const unsigned short* __restrict__ xT,
           const float* __restrict__ rowmax, const float* __restrict__ rowsum,
           float* __restrict__ out) {
  __shared__ float          ebuf[8][512];   // partial e: 2 tiles of 16x16 per wave
  __shared__ unsigned short pT[16 * 32];    // P^T tile, row stride 32 (n x m)
  const int b    = blockIdx.y;
  const int n0   = blockIdx.x * 16;
  const int tid  = threadIdx.x;
  const int wave = tid >> 5, lane = tid & 31;
  const int h    = lane >> 4, ln = lane & 15;
  const int dbase = wave * 128;

  v8f acc[8];
  #pragma unroll
  for (int t = 0; t < 8; ++t) acc[t] = vzero8();

  const unsigned short* brow = yb + ((size_t)b * N_ + n0 + ln) * D_;

  for (int m0 = 0; m0 < M_; m0 += 32) {
    // ---- Phase A: partial e tiles over this wave's d slice ----
    v8f e0 = vzero8(), e1 = vzero8();
    const unsigned short* a0row = xb + ((size_t)b * M_ + m0 + ln) * D_;
    const unsigned short* a1row = a0row + (size_t)16 * D_;
    #pragma unroll
    for (int kk = 0; kk < 128; kk += 32) {
      const int k0 = dbase + kk;
      Frag bm, a0, a1;
      bm.q[0] = *(const v4u*)(brow + k0 + h * 8);
      bm.q[1] = *(const v4u*)(brow + k0 + 16 + h * 8);
      a0.q[0] = *(const v4u*)(a0row + k0 + h * 8);
      a0.q[1] = *(const v4u*)(a0row + k0 + 16 + h * 8);
      a1.q[0] = *(const v4u*)(a1row + k0 + h * 8);
      a1.q[1] = *(const v4u*)(a1row + k0 + 16 + h * 8);
      e0 = wmma_bf16(a0, bm, e0);
      e1 = wmma_bf16(a1, bm, e1);
    }
    #pragma unroll
    for (int v = 0; v < 8; ++v) {
      int mrow = v + 8 * h;
      ebuf[wave][mrow * 16 + ln]       = e0[v];
      ebuf[wave][256 + mrow * 16 + ln] = e1[v];
    }
    __syncthreads();

    // ---- Phase B: reduce partials, softmax, write P^T (bf16) to LDS ----
    #pragma unroll
    for (int pass = 0; pass < 2; ++pass) {
      int idx = tid + pass * 256;
      float s = 0.f;
      #pragma unroll
      for (int w = 0; w < 8; ++w) s += ebuf[w][idx];
      int t  = idx >> 8;
      int mr = (idx >> 4) & 15;
      int nc = idx & 15;
      int gm = m0 + t * 16 + mr;
      float rm = rowmax[(size_t)b * M_ + gm];
      float rs = rowsum[(size_t)b * M_ + gm];
      float p  = __expf(s - rm) / rs;
      pT[nc * 32 + t * 16 + mr] = f2bf(p);
    }
    __syncthreads();

    // ---- Phase C: acc += P^T (A frag from LDS) x X (B frag from xT) ----
    Frag pa;
    pa.q[0] = *(const v4u*)(&pT[ln * 32 + h * 8]);
    pa.q[1] = *(const v4u*)(&pT[ln * 32 + 16 + h * 8]);
    #pragma unroll
    for (int t = 0; t < 8; ++t) {
      const int d = dbase + t * 16 + ln;
      const unsigned short* xtrow = xT + ((size_t)b * D_ + d) * M_ + m0;
      Frag bm;
      bm.q[0] = *(const v4u*)(xtrow + h * 8);
      bm.q[1] = *(const v4u*)(xtrow + 16 + h * 8);
      acc[t] = wmma_bf16(pa, bm, acc[t]);
    }
    __syncthreads();   // ebuf/pT reused next iteration
  }

  // ---- Epilogue: C layout -> out[b,n,d] ----
  #pragma unroll
  for (int t = 0; t < 8; ++t) {
    const int d = dbase + t * 16 + ln;
    #pragma unroll
    for (int v = 0; v < 8; ++v) {
      const int n = n0 + v + 8 * h;
      out[((size_t)b * N_ + n) * D_ + d] = acc[t][v];
    }
  }
}

// ---------------------------------------------------------------------------
extern "C" void kernel_launch(void* const* d_in, const int* in_sizes, int n_in,
                              void* d_out, int out_size, void* d_ws, size_t ws_size,
                              hipStream_t stream) {
  const float* x = (const float*)d_in[0];
  const float* y = (const float*)d_in[1];
  float* out = (float*)d_out;

  // Workspace layout (bytes):
  //   xb  : B*M*D*2 = 32 MiB  (bf16 x)
  //   yb  : B*N*D*2 = 32 MiB  (bf16 y)
  //   xT  : B*D*M*2 = 32 MiB  (bf16 x transposed, [b,d,m])
  //   rmax: B*M*4             (softmax row max)
  //   rsum: B*M*4             (softmax row denom)
  const size_t half_bytes = (size_t)B_ * M_ * D_ * 2;
  char* w = (char*)d_ws;
  unsigned short* xb = (unsigned short*)(w);
  unsigned short* yb = (unsigned short*)(w + half_bytes);
  unsigned short* xT = (unsigned short*)(w + 2 * half_bytes);
  float* rmax = (float*)(w + 3 * half_bytes);
  float* rsum = rmax + (size_t)B_ * M_;

  const int n4 = B_ * M_ * D_ / 4;   // float4 groups
  cvt_bf16_kernel<<<n4 / 256, 256, 0, stream>>>(x, xb, n4);
  cvt_bf16_kernel<<<n4 / 256, 256, 0, stream>>>(y, yb, n4);
  transpose_bf16_kernel<<<dim3(D_ / 32, M_ / 32, B_), 256, 0, stream>>>(x, xT);
  stats_kernel<<<dim3(M_ / 64, B_), 256, 0, stream>>>(xb, yb, rmax, rsum);
  out_kernel<<<dim3(N_ / 16, B_), 256, 0, stream>>>(xb, yb, xT, rmax, rsum, out);
}
